// transformerlayer_80513456931141
// MI455X (gfx1250) — matrix operand order repaired and checked by reading, not compiled, not run on hardware
//
#include <hip/hip_runtime.h>

// ---------------------------------------------------------------------------
// Transformer layer for MI455X (gfx1250, wave32, WMMA + TDM).
// B=4, S=1024, E=1024, H=16, D=64.  All matmuls via v_wmma_f32_16x16x32_f16.
// GEMMs: TDM (tensor_load_to_lds) double-buffered LDS staging, 8 waves/block.
// ---------------------------------------------------------------------------

#define E_ 1024
#define H_ 16
#define D_ 64
#define B_ 4
#define S_ 1024
#define M_ (B_ * S_)   // 4096 rows of activations

typedef __attribute__((ext_vector_type(16))) _Float16 v16h;
typedef __attribute__((ext_vector_type(8)))  _Float16 v8h;
typedef __attribute__((ext_vector_type(4)))  _Float16 v4h;
typedef __attribute__((ext_vector_type(8)))  float    v8f;
typedef __attribute__((ext_vector_type(4)))  unsigned int u32x4;
typedef __attribute__((ext_vector_type(8)))  int      i32x8;
typedef __attribute__((ext_vector_type(4)))  int      i32x4;

#if defined(__has_builtin)
#if __has_builtin(__builtin_amdgcn_tensor_load_to_lds) && \
    __has_builtin(__builtin_amdgcn_s_wait_tensorcnt)
#define USE_TDM 1
#endif
#endif
#ifndef USE_TDM
#define USE_TDM 0
#endif

__device__ __forceinline__ v8f zero8() {
  v8f z = {0.f, 0.f, 0.f, 0.f, 0.f, 0.f, 0.f, 0.f};
  return z;
}

// A-matrix 16x32 f16 layout (ISA 7.12.2): lane L (M = L&15), lhi = L>>4:
//   halves 0..7  hold K = lhi*8 + 0..7 ; halves 8..15 hold K = 16 + lhi*8 + 0..7
__device__ __forceinline__ v16h load_a_tile(const _Float16* rowp, int lhi) {
  v8h lo = *(const v8h*)(rowp + lhi * 8);
  v8h hi = *(const v8h*)(rowp + 16 + lhi * 8);
  v16h r;
#pragma unroll
  for (int j = 0; j < 8; ++j) { r[j] = lo[j]; r[8 + j] = hi[j]; }
  return r;
}

// B-matrix 32x16 f16 layout: lane L holds column N = L&15; halves 0..7 hold
// K = (L>>4)*8 + 0..7, halves 8..15 hold K = 16 + (L>>4)*8 + 0..7.
__device__ __forceinline__ v16h load_b_tile(const _Float16* colp, int lhi) {
  v8h lo = *(const v8h*)(colp + lhi * 8);
  v8h hi = *(const v8h*)(colp + 16 + lhi * 8);
  v16h r;
#pragma unroll
  for (int j = 0; j < 8; ++j) { r[j] = lo[j]; r[8 + j] = hi[j]; }
  return r;
}

__device__ __forceinline__ v8f wmma_f16(v16h a, v16h b, v8f c) {
  return __builtin_amdgcn_wmma_f32_16x16x32_f16(false, a, false, b,
                                                (short)0, c, false, false);
}

// ---------------------------------------------------------------------------
// GEMM epilogue (shared by both GEMM variants).
// EPI: 0 = f32 out, 1 = f16 out, 2 = f16 out transposed per head [B,H,D,S],
//      3 = +bias, quickGELU, f16 out, 4 = +bias, f32 out.
// ---------------------------------------------------------------------------
template <int EPI>
__device__ __forceinline__ void gemm_epilogue(
    v8f acc[2][4], int row0, int col0, int lhi, int lrow,
    const float* __restrict__ bias, float* __restrict__ outF,
    _Float16* __restrict__ outH, int N) {
#pragma unroll
  for (int mi = 0; mi < 2; ++mi)
#pragma unroll
    for (int ni = 0; ni < 4; ++ni)
#pragma unroll
      for (int r = 0; r < 8; ++r) {
        int row = row0 + mi * 16 + lhi * 8 + r;
        int col = col0 + ni * 16 + lrow;
        float v = acc[mi][ni][r];
        if (EPI == 0) {
          outF[(size_t)row * N + col] = v;
        } else if (EPI == 1) {
          outH[(size_t)row * N + col] = (_Float16)v;
        } else if (EPI == 2) {
          int b = row >> 10, s = row & (S_ - 1);
          int h = col >> 6,  d = col & (D_ - 1);
          outH[(((size_t)(b * H_ + h)) * D_ + d) * S_ + s] = (_Float16)v;
        } else if (EPI == 3) {
          v += bias[col];
          v = v / (1.f + __expf(-1.702f * v));   // quick GELU
          outH[(size_t)row * N + col] = (_Float16)v;
        } else {
          v += bias[col];
          outF[(size_t)row * N + col] = v;
        }
      }
}

#if USE_TDM
// ---------------------------------------------------------------------------
// TDM descriptor: load a 32(K) x 128(rows) f16 tile, row stride = K elements,
// into LDS with 4-DWORD padding after every 16 DWORDs (row stride 40 halves,
// 20-bank stride -> conflict-free fragment reads).
// D# layout per CDNA5 ISA 8.3/8.4.
// ---------------------------------------------------------------------------
#define LDSROW 40   // halves per LDS row (32 data + 8 pad)

__device__ __forceinline__ void tdm_load_tile(_Float16* lds_dst,
                                              const _Float16* gsrc,
                                              unsigned K, unsigned tensorRows) {
  unsigned long long ga = (unsigned long long)(size_t)gsrc;
  unsigned lds_off = (unsigned)(size_t)lds_dst;  // low 32 bits = LDS offset
  u32x4 g0;
  g0[0] = 1u;                                    // count=1 (valid), user mode
  g0[1] = lds_off;                               // lds_addr
  g0[2] = (unsigned)ga;                          // global_addr[31:0]
  g0[3] = (unsigned)((ga >> 32) & 0x01ffffffu) | (2u << 30);  // addr hi, type=2
  i32x8 g1;
  g1[0] = (int)((1u << 16)      // data_size = 2 bytes
              | (1u << 20)      // pad_enable
              | (3u << 22)      // pad_interval: 16 DWORDs
              | (3u << 25));    // pad_amount: 4 DWORDs
  g1[1] = (int)((K & 0xffffu) << 16);                        // tensor_dim0 lo16
  g1[2] = (int)((K >> 16) | ((tensorRows & 0xffffu) << 16)); // dim0 hi / dim1 lo
  g1[3] = (int)((tensorRows >> 16) | (32u << 16));           // dim1 hi, tile_dim0=32
  g1[4] = 128;                                               // tile_dim1 = 128 rows
  g1[5] = (int)K;                                            // dim0_stride lo32
  g1[6] = 0;
  g1[7] = 0;
  i32x4 zz = {0, 0, 0, 0};
#if defined(__clang_major__) && (__clang_major__ >= 23)
  // upstream clang-23 / therock-10.0 headers: 6-arg form
  i32x8 z8 = {0, 0, 0, 0, 0, 0, 0, 0};
  __builtin_amdgcn_tensor_load_to_lds(g0, g1, zz, zz, z8, 0);
#else
  // ROCm 7.2 (clang-22): 5-arg form
  __builtin_amdgcn_tensor_load_to_lds(g0, g1, zz, zz, 0);
#endif
}

// ---------------------------------------------------------------------------
// TDM + LDS double-buffered WMMA GEMM: C[M,N] = A[M,K] * Bt[N,K]^T.
// 256 threads (8 waves); block tile 128x128; wave tile 32x64; k-step 32.
// Wave 0 drives the TDM; TENSORcnt + one workgroup barrier per k-step.
// ---------------------------------------------------------------------------
template <int EPI>
__global__ __launch_bounds__(256) void gemm_wmma_tdm_kernel(
    const _Float16* __restrict__ A, const _Float16* __restrict__ Bt,
    const float* __restrict__ bias,
    float* __restrict__ outF, _Float16* __restrict__ outH,
    int M, int N, int K) {
  __shared__ __align__(16) _Float16 As[2][128 * LDSROW];
  __shared__ __align__(16) _Float16 Bs[2][128 * LDSROW];

  const int lane = threadIdx.x & 31;
  const int wave = threadIdx.x >> 5;
  const int lrow = lane & 15;
  const int lhi  = lane >> 4;
  const int m0 = blockIdx.y * 128;
  const int n0 = blockIdx.x * 128;
  const int wr = (wave & 3) * 32;   // wave row offset in block tile
  const int wc = (wave >> 2) * 64;  // wave col offset in block tile

  v8f acc[2][4];
#pragma unroll
  for (int i = 0; i < 2; ++i)
#pragma unroll
    for (int j = 0; j < 4; ++j) acc[i][j] = zero8();

  if (wave == 0) {
    tdm_load_tile(&As[0][0], A  + (size_t)m0 * K, (unsigned)K, (unsigned)M);
    tdm_load_tile(&Bs[0][0], Bt + (size_t)n0 * K, (unsigned)K, (unsigned)N);
  }

  const int nIter = K >> 5;
  for (int it = 0; it < nIter; ++it) {
    const int p = it & 1;
    if (wave == 0) __builtin_amdgcn_s_wait_tensorcnt(0);
    __syncthreads();  // buf p visible to all; all done reading buf p^1
    if (wave == 0 && (it + 1) < nIter) {
      tdm_load_tile(&As[p ^ 1][0], A  + (size_t)m0 * K + (it + 1) * 32,
                    (unsigned)K, (unsigned)M);
      tdm_load_tile(&Bs[p ^ 1][0], Bt + (size_t)n0 * K + (it + 1) * 32,
                    (unsigned)K, (unsigned)N);
    }
    v16h a0 = load_a_tile(&As[p][(wr + lrow) * LDSROW], lhi);
    v16h a1 = load_a_tile(&As[p][(wr + 16 + lrow) * LDSROW], lhi);
    v16h b0 = load_b_tile(&Bs[p][(wc + lrow) * LDSROW], lhi);
    v16h b1 = load_b_tile(&Bs[p][(wc + 16 + lrow) * LDSROW], lhi);
    v16h b2 = load_b_tile(&Bs[p][(wc + 32 + lrow) * LDSROW], lhi);
    v16h b3 = load_b_tile(&Bs[p][(wc + 48 + lrow) * LDSROW], lhi);
    acc[0][0] = wmma_f16(a0, b0, acc[0][0]);
    acc[0][1] = wmma_f16(a0, b1, acc[0][1]);
    acc[0][2] = wmma_f16(a0, b2, acc[0][2]);
    acc[0][3] = wmma_f16(a0, b3, acc[0][3]);
    acc[1][0] = wmma_f16(a1, b0, acc[1][0]);
    acc[1][1] = wmma_f16(a1, b1, acc[1][1]);
    acc[1][2] = wmma_f16(a1, b2, acc[1][2]);
    acc[1][3] = wmma_f16(a1, b3, acc[1][3]);
  }

  gemm_epilogue<EPI>(acc, m0 + wr, n0 + wc, lhi, lrow, bias, outF, outH, N);
}

#else  // !USE_TDM: direct-from-global fallback (4 waves, wave tile 32x64)

template <int EPI>
__global__ __launch_bounds__(128) void gemm_wmma_direct_kernel(
    const _Float16* __restrict__ A, const _Float16* __restrict__ Bt,
    const float* __restrict__ bias,
    float* __restrict__ outF, _Float16* __restrict__ outH,
    int M, int N, int K) {
  const int lane = threadIdx.x & 31;
  const int wave = threadIdx.x >> 5;
  const int lrow = lane & 15;
  const int lhi  = lane >> 4;
  const int row0 = blockIdx.y * 64 + (wave & 1) * 32;
  const int col0 = blockIdx.x * 128 + (wave >> 1) * 64;

  v8f acc[2][4];
#pragma unroll
  for (int i = 0; i < 2; ++i)
#pragma unroll
    for (int j = 0; j < 4; ++j) acc[i][j] = zero8();

  const _Float16* aRow0 = A + (size_t)(row0 + lrow) * K;
  const _Float16* aRow1 = aRow0 + (size_t)16 * K;
  const _Float16* bCol0 = Bt + (size_t)(col0 + lrow) * K;

  for (int k0 = 0; k0 < K; k0 += 32) {
    v16h a0 = load_a_tile(aRow0 + k0, lhi);
    v16h a1 = load_a_tile(aRow1 + k0, lhi);
    v16h b0 = load_b_tile(bCol0 + k0, lhi);
    v16h b1 = load_b_tile(bCol0 + (size_t)16 * K + k0, lhi);
    v16h b2 = load_b_tile(bCol0 + (size_t)32 * K + k0, lhi);
    v16h b3 = load_b_tile(bCol0 + (size_t)48 * K + k0, lhi);
    acc[0][0] = wmma_f16(a0, b0, acc[0][0]);
    acc[0][1] = wmma_f16(a0, b1, acc[0][1]);
    acc[0][2] = wmma_f16(a0, b2, acc[0][2]);
    acc[0][3] = wmma_f16(a0, b3, acc[0][3]);
    acc[1][0] = wmma_f16(a1, b0, acc[1][0]);
    acc[1][1] = wmma_f16(a1, b1, acc[1][1]);
    acc[1][2] = wmma_f16(a1, b2, acc[1][2]);
    acc[1][3] = wmma_f16(a1, b3, acc[1][3]);
  }
  gemm_epilogue<EPI>(acc, row0, col0, lhi, lrow, bias, outF, outH, N);
}
#endif

// ---------------------------------------------------------------------------
// f32 -> f16 conversion (4 elements/thread)
// ---------------------------------------------------------------------------
__global__ __launch_bounds__(256) void cvt_f16_kernel(
    const float* __restrict__ in, _Float16* __restrict__ out, int n) {
  int i = (blockIdx.x * 256 + threadIdx.x) * 4;
  if (i < n) {
    float4 v = *(const float4*)(in + i);
    v4h o = { (_Float16)v.x, (_Float16)v.y, (_Float16)v.z, (_Float16)v.w };
    *(v4h*)(out + i) = o;
  }
}

// ---------------------------------------------------------------------------
// Transpose + convert: in f32 [R,C] -> out f16 [C,R].  R,C multiples of 32.
// ---------------------------------------------------------------------------
__global__ __launch_bounds__(256) void transpose_cvt_kernel(
    const float* __restrict__ in, _Float16* __restrict__ out, int R, int C) {
  __shared__ float tile[32][33];
  int c = blockIdx.x * 32 + threadIdx.x;
  int r0 = blockIdx.y * 32;
#pragma unroll
  for (int j = 0; j < 32; j += 8)
    tile[threadIdx.y + j][threadIdx.x] = in[(size_t)(r0 + threadIdx.y + j) * C + c];
  __syncthreads();
  int oc  = r0 + threadIdx.x;
  int or0 = blockIdx.x * 32;
#pragma unroll
  for (int j = 0; j < 32; j += 8)
    out[(size_t)(or0 + threadIdx.y + j) * R + oc] =
        (_Float16)tile[threadIdx.x][threadIdx.y + j];
}

// ---------------------------------------------------------------------------
// Flash-style attention.  One wave = 16 query rows of one (b,h).
// q16,k16: [B,S,E] f16;  vT16: [B,H,D,S] f16;  out16: [B,S,E] f16.
// grid = (S/64, H, B), block = 128 (4 waves).
// ---------------------------------------------------------------------------
__global__ __launch_bounds__(128) void attention_kernel(
    const _Float16* __restrict__ q16, const _Float16* __restrict__ k16,
    const _Float16* __restrict__ vT16,
    const unsigned char* __restrict__ mask_q,
    const unsigned char* __restrict__ mask_k,
    _Float16* __restrict__ out16) {
  __shared__ __align__(32) _Float16 psh[4][16][32];
  const int lane = threadIdx.x & 31;
  const int wave = threadIdx.x >> 5;
  const int lrow = lane & 15;
  const int lhi  = lane >> 4;
  const int b = blockIdx.z, h = blockIdx.y;
  const int q0 = blockIdx.x * 64 + wave * 16;

  const _Float16* qrow = q16 + ((size_t)(b * S_ + q0 + lrow)) * E_ + h * D_;
  v16h aq0 = load_a_tile(qrow, lhi);        // d = 0..31
  v16h aq1 = load_a_tile(qrow + 32, lhi);   // d = 32..63

  unsigned rmask = 0;
#pragma unroll
  for (int r = 0; r < 8; ++r)
    rmask |= (mask_q[b * S_ + q0 + lhi * 8 + r] ? 1u : 0u) << r;

  float m[8], l[8];
  v8f acc[4];
#pragma unroll
  for (int r = 0; r < 8; ++r) { m[r] = -3.0e38f; l[r] = 0.f; }
#pragma unroll
  for (int t = 0; t < 4; ++t) acc[t] = zero8();

  const float scale = 0.03125f;  // 1/sqrt(E)

  for (int n0 = 0; n0 < S_; n0 += 32) {
    const _Float16* kc0 = k16 + ((size_t)(b * S_ + n0 + lrow)) * E_ + h * D_;
    const _Float16* kc1 = kc0 + (size_t)16 * E_;
    v8f e0 = zero8(), e1 = zero8();
    e0 = wmma_f16(aq0, load_b_tile(kc0,      lhi), e0);
    e0 = wmma_f16(aq1, load_b_tile(kc0 + 32, lhi), e0);
    e1 = wmma_f16(aq0, load_b_tile(kc1,      lhi), e1);
    e1 = wmma_f16(aq1, load_b_tile(kc1 + 32, lhi), e1);

    const bool km0 = mask_k[b * S_ + n0 + lrow] != 0;
    const bool km1 = mask_k[b * S_ + n0 + 16 + lrow] != 0;

    float p0[8], p1[8];
#pragma unroll
    for (int r = 0; r < 8; ++r) {
      bool qm = (rmask >> r) & 1;
      float x0 = (qm && km0) ? e0[r] * scale : -1.0e30f;
      float x1 = (qm && km1) ? e1[r] * scale : -1.0e30f;
      float t = fmaxf(x0, x1);
#pragma unroll
      for (int off = 1; off < 16; off <<= 1) t = fmaxf(t, __shfl_xor(t, off, 32));
      float nm = fmaxf(m[r], t);
      float f = __expf(m[r] - nm);
      p0[r] = __expf(x0 - nm);
      p1[r] = __expf(x1 - nm);
      float s = p0[r] + p1[r];
#pragma unroll
      for (int off = 1; off < 16; off <<= 1) s += __shfl_xor(s, off, 32);
      l[r] = l[r] * f + s;
      m[r] = nm;
#pragma unroll
      for (int t2 = 0; t2 < 4; ++t2) acc[t2][r] = acc[t2][r] * f;
    }

    // bounce P (C-layout) through LDS to re-read it in A-layout
#pragma unroll
    for (int r = 0; r < 8; ++r) {
      psh[wave][lhi * 8 + r][lrow]      = (_Float16)p0[r];
      psh[wave][lhi * 8 + r][16 + lrow] = (_Float16)p1[r];
    }
    asm volatile("s_wait_dscnt 0" ::: "memory");  // wave-internal LDS RAW
    v16h ap = load_a_tile(&psh[wave][lrow][0], lhi);

    const _Float16* vrow =
        vT16 + ((size_t)((b * H_ + h) * D_ + lrow)) * S_ + n0;
#pragma unroll
    for (int t2 = 0; t2 < 4; ++t2)
      acc[t2] = wmma_f16(ap, load_b_tile(vrow + (size_t)t2 * 16 * S_, lhi), acc[t2]);
  }

#pragma unroll
  for (int t2 = 0; t2 < 4; ++t2)
#pragma unroll
    for (int r = 0; r < 8; ++r) {
      int row = q0 + lhi * 8 + r;
      int col = h * D_ + t2 * 16 + lrow;
      float v = acc[t2][r] / l[r];
      if (!((rmask >> r) & 1)) v = 0.f;
      out16[((size_t)(b * S_ + row)) * E_ + col] = (_Float16)v;
    }
}

// ---------------------------------------------------------------------------
// Block reduction helper (256 threads, 8 waves)
// ---------------------------------------------------------------------------
__device__ __forceinline__ float block_reduce_sum(float v, float* sh) {
#pragma unroll
  for (int off = 16; off > 0; off >>= 1) v += __shfl_xor(v, off, 32);
  __syncthreads();
  if ((threadIdx.x & 31) == 0) sh[threadIdx.x >> 5] = v;
  __syncthreads();
  float total = 0.f;
#pragma unroll
  for (int i = 0; i < 8; ++i) total += sh[i];
  return total;
}

// LN0: h16 = (mask_q ? LN(att)*g+b : 0) + query   (f16 out, feeds FFN1)
__global__ __launch_bounds__(256) void ln0_kernel(
    const float* __restrict__ att, const float* __restrict__ query,
    const unsigned char* __restrict__ mq,
    const float* __restrict__ g, const float* __restrict__ bb,
    _Float16* __restrict__ h16) {
  __shared__ float sh[8];
  int row = blockIdx.x;
  const float* x = att + (size_t)row * E_;
  float v[4], s = 0.f;
#pragma unroll
  for (int i = 0; i < 4; ++i) { v[i] = x[threadIdx.x + i * 256]; s += v[i]; }
  float mu = block_reduce_sum(s, sh) * (1.f / E_);
  float s2 = 0.f;
#pragma unroll
  for (int i = 0; i < 4; ++i) { float d = v[i] - mu; s2 += d * d; }
  float var = block_reduce_sum(s2, sh) * (1.f / E_);
  float rs = rsqrtf(var + 1e-5f);
  bool qm = mq[row] != 0;
#pragma unroll
  for (int i = 0; i < 4; ++i) {
    int col = threadIdx.x + i * 256;
    float y = (v[i] - mu) * rs * g[col] + bb[col];
    if (!qm) y = 0.f;
    y += query[(size_t)row * E_ + col];
    h16[(size_t)row * E_ + col] = (_Float16)y;
  }
}

// LN1: out = LN(ff2 + query)*g+b, then * mask_q   (f32 out = final result)
__global__ __launch_bounds__(256) void ln1_kernel(
    const float* __restrict__ ff2, const float* __restrict__ query,
    const unsigned char* __restrict__ mq,
    const float* __restrict__ g, const float* __restrict__ bb,
    float* __restrict__ out) {
  __shared__ float sh[8];
  int row = blockIdx.x;
  float v[4], s = 0.f;
#pragma unroll
  for (int i = 0; i < 4; ++i) {
    int col = threadIdx.x + i * 256;
    v[i] = ff2[(size_t)row * E_ + col] + query[(size_t)row * E_ + col];
    s += v[i];
  }
  float mu = block_reduce_sum(s, sh) * (1.f / E_);
  float s2 = 0.f;
#pragma unroll
  for (int i = 0; i < 4; ++i) { float d = v[i] - mu; s2 += d * d; }
  float var = block_reduce_sum(s2, sh) * (1.f / E_);
  float rs = rsqrtf(var + 1e-5f);
  float qm = (mq[row] != 0) ? 1.f : 0.f;
#pragma unroll
  for (int i = 0; i < 4; ++i) {
    int col = threadIdx.x + i * 256;
    out[(size_t)row * E_ + col] = ((v[i] - mu) * rs * g[col] + bb[col]) * qm;
  }
}

// ---------------------------------------------------------------------------
// Host-side GEMM launcher (selects TDM vs direct variant)
// ---------------------------------------------------------------------------
template <int EPI>
static void launch_gemm(const _Float16* A, const _Float16* Bt, const float* bias,
                        float* outF, _Float16* outH, int M, int N, int K,
                        hipStream_t s) {
#if USE_TDM
  gemm_wmma_tdm_kernel<EPI><<<dim3(N / 128, M / 128), 256, 0, s>>>(
      A, Bt, bias, outF, outH, M, N, K);
#else
  gemm_wmma_direct_kernel<EPI><<<dim3(N / 128, M / 64), 128, 0, s>>>(
      A, Bt, bias, outF, outH, M, N, K);
#endif
}

// ---------------------------------------------------------------------------
// Orchestration.  Workspace overlay (64 MiB total):
//   [ 0, 8)MiB  val16 -> attn16 -> h16
//   [ 8,24)MiB  key16/qry16 -> att32 -> ff232
//   [24,40)MiB  transposed f16 weights (live whole pass)
//   [40,64)MiB  vT16/k16/q16 -> ff116
// ---------------------------------------------------------------------------
extern "C" void kernel_launch(void* const* d_in, const int* in_sizes, int n_in,
                              void* d_out, int out_size, void* d_ws, size_t ws_size,
                              hipStream_t stream) {
  const float* value  = (const float*)d_in[0];
  const float* key_   = (const float*)d_in[1];
  const float* query  = (const float*)d_in[2];
  const unsigned char* mask_k = (const unsigned char*)d_in[3];
  const unsigned char* mask_q = (const unsigned char*)d_in[4];
  const float* Wv = (const float*)d_in[5];
  const float* Wk = (const float*)d_in[6];
  const float* Wq = (const float*)d_in[7];
  const float* Wo = (const float*)d_in[8];
  const float* ln0_g = (const float*)d_in[9];
  const float* ln0_b = (const float*)d_in[10];
  const float* W1 = (const float*)d_in[11];
  const float* b1 = (const float*)d_in[12];
  const float* W2 = (const float*)d_in[13];
  const float* b2 = (const float*)d_in[14];
  const float* ln1_g = (const float*)d_in[15];
  const float* ln1_b = (const float*)d_in[16];

  char* ws = (char*)d_ws;
  const size_t MB = (size_t)1 << 20;
  _Float16* val16 = (_Float16*)(ws + 0 * MB);
  _Float16* key16 = (_Float16*)(ws + 8 * MB);
  _Float16* qry16 = (_Float16*)(ws + 16 * MB);
  _Float16* Wvt   = (_Float16*)(ws + 24 * MB);
  _Float16* Wkt   = (_Float16*)(ws + 26 * MB);
  _Float16* Wqt   = (_Float16*)(ws + 28 * MB);
  _Float16* Wot   = (_Float16*)(ws + 30 * MB);
  _Float16* W1t   = (_Float16*)(ws + 32 * MB);
  _Float16* W2t   = (_Float16*)(ws + 36 * MB);
  _Float16* vT16  = (_Float16*)(ws + 40 * MB);
  _Float16* k16   = (_Float16*)(ws + 48 * MB);
  _Float16* q16   = (_Float16*)(ws + 56 * MB);
  _Float16* attn16 = (_Float16*)(ws + 0 * MB);   // reuse val16
  float*    att32  = (float*)(ws + 8 * MB);      // reuse key16+qry16
  _Float16* h16    = (_Float16*)(ws + 0 * MB);   // reuse attn16
  _Float16* ff116  = (_Float16*)(ws + 40 * MB);  // reuse vT/k/q
  float*    ff232  = (float*)(ws + 8 * MB);      // reuse att32

  const int nME = M_ * E_;
  cvt_f16_kernel<<<nME / 1024, 256, 0, stream>>>(value, val16, nME);
  cvt_f16_kernel<<<nME / 1024, 256, 0, stream>>>(key_,  key16, nME);
  cvt_f16_kernel<<<nME / 1024, 256, 0, stream>>>(query, qry16, nME);

  dim3 tb(32, 8);
  transpose_cvt_kernel<<<dim3(E_ / 32, E_ / 32), tb, 0, stream>>>(Wv, Wvt, E_, E_);
  transpose_cvt_kernel<<<dim3(E_ / 32, E_ / 32), tb, 0, stream>>>(Wk, Wkt, E_, E_);
  transpose_cvt_kernel<<<dim3(E_ / 32, E_ / 32), tb, 0, stream>>>(Wq, Wqt, E_, E_);
  transpose_cvt_kernel<<<dim3(E_ / 32, E_ / 32), tb, 0, stream>>>(Wo, Wot, E_, E_);
  transpose_cvt_kernel<<<dim3(2 * E_ / 32, E_ / 32), tb, 0, stream>>>(W1, W1t, E_, 2 * E_);
  transpose_cvt_kernel<<<dim3(E_ / 32, 2 * E_ / 32), tb, 0, stream>>>(W2, W2t, 2 * E_, E_);

  launch_gemm<2>(val16, Wvt, nullptr, nullptr, vT16, M_, E_, E_, stream);
  launch_gemm<1>(key16, Wkt, nullptr, nullptr, k16,  M_, E_, E_, stream);
  launch_gemm<1>(qry16, Wqt, nullptr, nullptr, q16,  M_, E_, E_, stream);

  attention_kernel<<<dim3(S_ / 64, H_, B_), 128, 0, stream>>>(
      q16, k16, vT16, mask_q, mask_k, attn16);

  launch_gemm<0>(attn16, Wot, nullptr, att32, nullptr, M_, E_, E_, stream);
  ln0_kernel<<<M_, 256, 0, stream>>>(att32, query, mask_q, ln0_g, ln0_b, h16);

  launch_gemm<3>(h16, W1t, b1, nullptr, ff116, M_, 2 * E_, E_, stream);
  launch_gemm<4>(ff116, W2t, b2, ff232, nullptr, M_, E_, 2 * E_, stream);

  ln1_kernel<<<M_, 256, 0, stream>>>(ff232, query, mask_q, ln1_g, ln1_b, (float*)d_out);
}
